// AttnBlockpp_7619271983716
// MI455X (gfx1250) — matrix operand order repaired and checked by reading, not compiled, not run on hardware
//
#include <hip/hip_runtime.h>
#include <hip/hip_bf16.h>
#include <stdint.h>

typedef __attribute__((ext_vector_type(16))) __bf16 v16bf;
typedef __attribute__((ext_vector_type(8)))  float  v8f;
typedef uint32_t u32x4 __attribute__((ext_vector_type(4)));
typedef uint32_t u32x8 __attribute__((ext_vector_type(8)));

#define HW   4096
#define C    256
#define BATCH 2
#define NH   4
#define CH   64
#define G    32
#define CPG  8
#define MTOT (BATCH*HW)   // 8192

__device__ inline unsigned short f2bf(float f) {
  union { float f; unsigned int u; } x; x.f = f;
  unsigned int u = x.u;
  return (unsigned short)((u + 0x7FFFu + ((u >> 16) & 1u)) >> 16);
}

union FragB { int i[8]; unsigned short us[16]; v16bf v; };

// ---- Tensor Data Mover: 2-D tile (16-bit elements) global -> LDS -------------
// D# group0/group1 packed per CDNA5 ISA §8.3/8.4; 2-operand form (VADDR2/3=NULL,
// tensors up to 2D). Issued once per workgroup by wave 0; tracked by TENSORcnt.
__device__ inline void tdm_load_2d_b16(uint32_t lds_off, const void* gptr,
                                       uint32_t tile_w, uint32_t tile_h,
                                       uint32_t stride_elems) {
  uint64_t ga = (uint64_t)(uintptr_t)gptr;
  u32x4 g0;
  g0[0] = 1u;                                  // count=1, user descriptor
  g0[1] = lds_off;                             // lds_addr (bytes)
  g0[2] = (uint32_t)ga;                        // global_addr[31:0]
  g0[3] = (uint32_t)(ga >> 32) | (2u << 30);   // global_addr[56:32] | type=2
  u32x8 g1;
  g1[0] = 1u << 16;                            // workgroup_mask=0, data_size=2B
  g1[1] = (stride_elems & 0xFFFFu) << 16;      // tensor_dim0[15:0]
  g1[2] = (stride_elems >> 16) | ((tile_h & 0xFFFFu) << 16);  // td0 hi | tensor_dim1 lo
  g1[3] = (tile_h >> 16) | (tile_w << 16);     // tensor_dim1 hi | tile_dim0
  g1[4] = tile_h;                              // tile_dim1 (tile_dim2 = 0 -> 2D)
  g1[5] = stride_elems;                        // tensor_dim0_stride[31:0]
  g1[6] = 0u;                                  // stride hi / dim1_stride lo
  g1[7] = 0u;
  asm volatile("tensor_load_to_lds %0, %1" :: "s"(g0), "s"(g1) : "memory");
}

__device__ inline void wait_tensor0() {
#if __has_builtin(__builtin_amdgcn_s_wait_tensorcnt)
  __builtin_amdgcn_s_wait_tensorcnt((short)0);
#else
  asm volatile("s_wait_tensorcnt 0x0" ::: "memory");
#endif
}

// ---------------------------------------------------------------- GroupNorm
__global__ __launch_bounds__(256) void gn_stats_kernel(const float* __restrict__ x,
                                                       float* __restrict__ stats) {
  int bg = blockIdx.x;                              // (b*32+g), 0..63
  const float* p = x + (size_t)bg * CPG * HW;       // group = 32768 contiguous floats
  float s = 0.f, s2 = 0.f;
  for (int i = threadIdx.x; i < CPG * HW; i += 256) {
    float v = p[i]; s += v; s2 += v * v;
  }
  for (int off = 16; off; off >>= 1) {
    s  += __shfl_xor(s,  off, 32);
    s2 += __shfl_xor(s2, off, 32);
  }
  __shared__ float ls[8], ls2[8];
  int w = threadIdx.x >> 5;
  if ((threadIdx.x & 31) == 0) { ls[w] = s; ls2[w] = s2; }
  __syncthreads();
  if (threadIdx.x == 0) {
    float ts = 0.f, ts2 = 0.f;
    for (int i = 0; i < 8; i++) { ts += ls[i]; ts2 += ls2[i]; }
    float inv = 1.f / (float)(CPG * HW);
    float mean = ts * inv;
    float var = ts2 * inv - mean * mean;
    stats[2 * bg]     = mean;
    stats[2 * bg + 1] = rsqrtf(var + 1e-6f);
  }
}

// normalize + transpose [b,c,pos] fp32 -> [b,pos,c] bf16
__global__ __launch_bounds__(256) void gn_apply_kernel(const float* __restrict__ x,
                                                       const float* __restrict__ gscale,
                                                       const float* __restrict__ gbias,
                                                       const float* __restrict__ stats,
                                                       unsigned short* __restrict__ hbf) {
  __shared__ float tile[32][33];
  int p0 = blockIdx.x * 32, c0 = blockIdx.y * 32, b = blockIdx.z;
  int tj = threadIdx.x & 31;
  int ti = threadIdx.x >> 5;
  for (int rr = 0; rr < 4; rr++) {
    int cl = ti + rr * 8;
    int c = c0 + cl;
    int g = c >> 3;
    float mean = stats[2 * (b * G + g)];
    float rstd = stats[2 * (b * G + g) + 1];
    float v = x[((size_t)(b * C + c)) * HW + p0 + tj];
    tile[cl][tj] = (v - mean) * rstd * gscale[c] + gbias[c];
  }
  __syncthreads();
  for (int rr = 0; rr < 4; rr++) {
    int pl = ti + rr * 8;
    hbf[((size_t)(b * HW + p0 + pl)) * C + c0 + tj] = f2bf(tile[tj][pl]);
  }
}

// ---------------------------------------------------------------- weights -> bf16
__global__ __launch_bounds__(256) void wconv_kernel(const float* __restrict__ W0,
                                                    const float* __restrict__ W1,
                                                    const float* __restrict__ W2,
                                                    const float* __restrict__ W3,
                                                    unsigned short* __restrict__ wbf) {
  int i = blockIdx.x * 256 + threadIdx.x;
  int sel = i >> 16, j = i & 65535;
  const float* src = sel == 0 ? W0 : sel == 1 ? W1 : sel == 2 ? W2 : W3;
  wbf[i] = f2bf(src[j]);
}

// ---------------------------------------------------------------- GEMM (M=8192,N=256,K=256)
// A tile staged by TDM (wave 0); B tile transposed manually by all waves.
// mode 0: out = bf16 q/k/v split per head [(b*NH+h)][pos][64]
// mode 1: out = fp32 (x + A*W + bias) / sqrt(2), layout [b][c][pos]
__global__ __launch_bounds__(256) void gemm_kernel(const unsigned short* __restrict__ A,
                                                   const unsigned short* __restrict__ Wb,
                                                   const float* __restrict__ bias,
                                                   unsigned short* __restrict__ outq,
                                                   const float* __restrict__ xres,
                                                   float* __restrict__ outf,
                                                   int mode) {
  __shared__ unsigned short Asm[128 * 32];
  __shared__ unsigned short Bsm[128 * 36];     // transposed [n][k], padded stride
  int MB = blockIdx.x * 128, NB = blockIdx.y * 128;
  int t = threadIdx.x;
  int lane = t & 31, w = t >> 5;
  int wr = w >> 1, wc = w & 1;                 // 4x2 waves, wave tile 32x64
  int g = lane >> 4, l16 = lane & 15;
  int wuni = __builtin_amdgcn_readfirstlane(w);      // wave-uniform wave id
  uint32_t asm_off = (uint32_t)(uintptr_t)&Asm[0];   // LDS byte offset

  v8f acc[2][4];
  for (int i = 0; i < 2; i++)
    for (int j = 0; j < 4; j++) acc[i][j] = (v8f)(0.f);

  for (int kk = 0; kk < C; kk += 32) {
    // async: DMA the 128x32 A tile into LDS (wave 0 only, TENSORcnt-tracked)
    if (wuni == 0)
      tdm_load_2d_b16(asm_off, A + (size_t)MB * C + kk, /*w*/32, /*h*/128, /*stride*/C);
    // meanwhile: stage B transposed: Bsm[n][k] = W[kk+k][NB+n]
    {
      int kr = t >> 3, nseg = (t & 7) * 16;
      union { uint4 q; unsigned short us[8]; } d0, d1;
      d0.q = *(const uint4*)(Wb + (size_t)(kk + kr) * C + NB + nseg);
      d1.q = *(const uint4*)(Wb + (size_t)(kk + kr) * C + NB + nseg + 8);
      for (int i = 0; i < 8; i++) Bsm[(nseg + i) * 36 + kr]     = d0.us[i];
      for (int i = 0; i < 8; i++) Bsm[(nseg + 8 + i) * 36 + kr] = d1.us[i];
    }
    if (wuni == 0) wait_tensor0();
    __syncthreads();

    FragB af[2];
    const int* LA = (const int*)Asm;
    for (int rt = 0; rt < 2; rt++) {
      int m = wr * 32 + rt * 16 + l16;
      int o = m * 16 + 4 * g;                  // elements 0..7: K=8g.., 8..15: K=8g+16..
      for (int i = 0; i < 4; i++) af[rt].i[i]     = LA[o + i];
      for (int i = 0; i < 4; i++) af[rt].i[4 + i] = LA[o + 8 + i];
    }
    const int* LB = (const int*)Bsm;
    for (int ct = 0; ct < 4; ct++) {
      FragB bfr;
      int n = wc * 64 + ct * 16 + l16;
      int o = n * 18 + 4 * g;                  // elements 0..7: K=8g.., 8..15: K=8g+16..
      for (int i = 0; i < 4; i++) bfr.i[i]     = LB[o + i];
      for (int i = 0; i < 4; i++) bfr.i[4 + i] = LB[o + 8 + i];
      for (int rt = 0; rt < 2; rt++)
        acc[rt][ct] = __builtin_amdgcn_wmma_f32_16x16x32_bf16(
            false, af[rt].v, false, bfr.v, (short)0, acc[rt][ct], false, false);
    }
    __syncthreads();
  }

  for (int rt = 0; rt < 2; rt++)
    for (int ct = 0; ct < 4; ct++) {
      int n = NB + wc * 64 + ct * 16 + l16;
      float bv = bias[n];
      for (int r = 0; r < 8; r++) {
        int m = MB + wr * 32 + rt * 16 + r + 8 * g;
        float val = acc[rt][ct][r] + bv;
        int b = m >> 12, pos = m & 4095;
        if (mode == 0) {
          int head = n >> 6, d = n & 63;
          outq[(((size_t)(b * NH + head)) * HW + pos) * CH + d] = f2bf(val);
        } else {
          size_t oi = ((size_t)(b * C + n)) * HW + pos;
          outf[oi] = (xres[oi] + val) * 0.70710678118654752f;
        }
      }
    }
}

// ---------------------------------------------------------------- flash attention
// grid (HW/128, B*NH); each wave: 16 q-rows x full 64-col K blocks, online softmax.
// K block DMA'd by TDM (wave 0); V block transposed manually by all waves.
__global__ __launch_bounds__(256) void flash_kernel(const unsigned short* __restrict__ q,
                                                    const unsigned short* __restrict__ k,
                                                    const unsigned short* __restrict__ v,
                                                    unsigned short* __restrict__ attn) {
  __shared__ unsigned short Kl[64 * 64];   // K block, natural [j][c] layout == B-frag layout
  __shared__ unsigned short Vt[64 * 72];   // V block transposed [d][j], padded
  __shared__ unsigned short Pl[128 * 64];  // per-wave P staging (A-frag layout source)
  int bh = blockIdx.y;
  int blockM = blockIdx.x * 128;
  int t = threadIdx.x, lane = t & 31, w = t >> 5;
  int g = lane >> 4, l16 = lane & 15;
  int wuni = __builtin_amdgcn_readfirstlane(w);
  uint32_t kl_off = (uint32_t)(uintptr_t)&Kl[0];
  const unsigned short* qp = q + (size_t)bh * HW * CH;
  const unsigned short* kp = k + (size_t)bh * HW * CH;
  const unsigned short* vp = v + (size_t)bh * HW * CH;

  FragB qf[2];                               // Q A-fragments pinned in registers
  {
    int m = blockM + w * 16 + l16;
    for (int cc = 0; cc < 2; cc++) {
      uint4 d0 = *(const uint4*)(qp + (size_t)m * CH + cc * 32 + 8 * g);
      uint4 d1 = *(const uint4*)(qp + (size_t)m * CH + cc * 32 + 8 * g + 16);
      *(uint4*)&qf[cc].i[0] = d0;
      *(uint4*)&qf[cc].i[4] = d1;
    }
  }

  float mrow[8], lrow[8];
  v8f acc[4];
  for (int i = 0; i < 8; i++) { mrow[i] = -1e30f; lrow[i] = 0.f; }
  for (int i = 0; i < 4; i++) acc[i] = (v8f)(0.f);
  const float scale = 0.125f;                // Ch^-0.5 = 1/8

  for (int kb = 0; kb < HW; kb += 64) {
    // async: DMA 64x64 K block into LDS (wave 0)
    if (wuni == 0)
      tdm_load_2d_b16(kl_off, kp + (size_t)kb * CH, /*w*/CH, /*h*/64, /*stride*/CH);
    // prefetch next V block into caches while we work on this one
    if (kb + 64 < HW)
      __builtin_prefetch(vp + (size_t)(kb + 64) * CH + t * 16, 0, 3);
    {   // stage V transposed (all waves, overlapped with TDM)
      int row = t >> 2, seg = (t & 3) * 16;
      union { uint4 u; unsigned short us[8]; } e0, e1;
      e0.u = *(const uint4*)(vp + (size_t)(kb + row) * CH + seg);
      e1.u = *(const uint4*)(vp + (size_t)(kb + row) * CH + seg + 8);
      for (int i = 0; i < 8; i++) Vt[(seg + i) * 72 + row]     = e0.us[i];
      for (int i = 0; i < 8; i++) Vt[(seg + 8 + i) * 72 + row] = e1.us[i];
    }
    if (wuni == 0) wait_tensor0();
    __syncthreads();

    // S = (Q K^T) * scale
    v8f s[4];
    const int* LK = (const int*)Kl;
    for (int ct = 0; ct < 4; ct++) {
      s[ct] = (v8f)(0.f);
      for (int cc = 0; cc < 2; cc++) {
        FragB kf;
        int o = (ct * 16 + l16) * 32 + cc * 16 + 4 * g;
        for (int i = 0; i < 4; i++) kf.i[i]     = LK[o + i];
        for (int i = 0; i < 4; i++) kf.i[4 + i] = LK[o + 8 + i];
        s[ct] = __builtin_amdgcn_wmma_f32_16x16x32_bf16(
            false, qf[cc].v, false, kf.v, (short)0, s[ct], false, false);
      }
    }

    // online softmax: rows live within a 16-lane half (M = r + 8g)
    float mnew[8];
    for (int r = 0; r < 8; r++) {
      float mx = -1e30f;
      for (int ct = 0; ct < 4; ct++) { s[ct][r] *= scale; mx = fmaxf(mx, s[ct][r]); }
      for (int off = 1; off < 16; off <<= 1) mx = fmaxf(mx, __shfl_xor(mx, off, 32));
      mnew[r] = fmaxf(mrow[r], mx);
    }
    unsigned short* Pw = Pl + w * 16 * 64;
    float rs[8];
    for (int r = 0; r < 8; r++) rs[r] = 0.f;
    for (int ct = 0; ct < 4; ct++)
      for (int r = 0; r < 8; r++) {
        float p = __expf(s[ct][r] - mnew[r]);
        rs[r] += p;
        Pw[(r + 8 * g) * 64 + ct * 16 + l16] = f2bf(p);   // C-layout -> A-layout via LDS
      }
    for (int r = 0; r < 8; r++) {
      for (int off = 1; off < 16; off <<= 1) rs[r] += __shfl_xor(rs[r], off, 32);
      float alpha = __expf(mrow[r] - mnew[r]);
      lrow[r] = lrow[r] * alpha + rs[r];
      mrow[r] = mnew[r];
      for (int ct = 0; ct < 4; ct++) acc[ct][r] *= alpha;
    }

    // O += P * V   (same-wave LDS ops are in-order; cross-wave isolation by region)
    const int* LP = (const int*)Pw;
    const int* LV = (const int*)Vt;
    FragB pf[2];
    for (int jc = 0; jc < 2; jc++) {
      int o = l16 * 32 + jc * 16 + 4 * g;
      for (int i = 0; i < 4; i++) pf[jc].i[i]     = LP[o + i];
      for (int i = 0; i < 4; i++) pf[jc].i[4 + i] = LP[o + 8 + i];
    }
    for (int ct = 0; ct < 4; ct++)
      for (int jc = 0; jc < 2; jc++) {
        FragB vf;
        int o = (ct * 16 + l16) * 36 + jc * 16 + 4 * g;
        for (int i = 0; i < 4; i++) vf.i[i]     = LV[o + i];
        for (int i = 0; i < 4; i++) vf.i[4 + i] = LV[o + 8 + i];
        acc[ct] = __builtin_amdgcn_wmma_f32_16x16x32_bf16(
            false, pf[jc].v, false, vf.v, (short)0, acc[ct], false, false);
      }
    __syncthreads();
  }

  int b = bh >> 2, head = bh & 3;
  for (int ct = 0; ct < 4; ct++)
    for (int r = 0; r < 8; r++) {
      int pos = blockM + w * 16 + r + 8 * g;
      int c = head * 64 + ct * 16 + l16;
      attn[((size_t)(b * HW + pos)) * C + c] = f2bf(acc[ct][r] / lrow[r]);
    }
}

// ---------------------------------------------------------------- launcher
extern "C" void kernel_launch(void* const* d_in, const int* in_sizes, int n_in,
                              void* d_out, int out_size, void* d_ws, size_t ws_size,
                              hipStream_t stream) {
  (void)in_sizes; (void)n_in; (void)out_size; (void)ws_size;
  const float* x  = (const float*)d_in[0];
  const float* gs = (const float*)d_in[1];
  const float* gb = (const float*)d_in[2];
  const float* W0 = (const float*)d_in[3];
  const float* b0 = (const float*)d_in[4];
  const float* W1 = (const float*)d_in[5];
  const float* b1 = (const float*)d_in[6];
  const float* W2 = (const float*)d_in[7];
  const float* b2 = (const float*)d_in[8];
  const float* W3 = (const float*)d_in[9];
  const float* b3 = (const float*)d_in[10];
  float* out = (float*)d_out;

  char* ws = (char*)d_ws;
  size_t off = 0;
  auto alloc = [&](size_t bytes) {
    size_t o = off;
    off = (off + bytes + 255) & ~(size_t)255;
    return o;
  };
  float* stats          = (float*)(ws + alloc(64 * 2 * sizeof(float)));
  unsigned short* hbf   = (unsigned short*)(ws + alloc((size_t)MTOT * C * 2));
  unsigned short* wbf   = (unsigned short*)(ws + alloc((size_t)4 * C * C * 2));
  unsigned short* qq    = (unsigned short*)(ws + alloc((size_t)MTOT * C * 2));
  unsigned short* kk    = (unsigned short*)(ws + alloc((size_t)MTOT * C * 2));
  unsigned short* vv    = (unsigned short*)(ws + alloc((size_t)MTOT * C * 2));
  unsigned short* attn  = (unsigned short*)(ws + alloc((size_t)MTOT * C * 2));

  wconv_kernel<<<4 * C * C / 256, 256, 0, stream>>>(W0, W1, W2, W3, wbf);
  gn_stats_kernel<<<BATCH * G, 256, 0, stream>>>(x, stats);
  gn_apply_kernel<<<dim3(HW / 32, C / 32, BATCH), 256, 0, stream>>>(x, gs, gb, stats, hbf);

  dim3 gg(MTOT / 128, C / 128);
  gemm_kernel<<<gg, 256, 0, stream>>>(hbf, wbf + 0 * C * C, b0, qq, nullptr, nullptr, 0);
  gemm_kernel<<<gg, 256, 0, stream>>>(hbf, wbf + 1 * C * C, b1, kk, nullptr, nullptr, 0);
  gemm_kernel<<<gg, 256, 0, stream>>>(hbf, wbf + 2 * C * C, b2, vv, nullptr, nullptr, 0);

  flash_kernel<<<dim3(HW / 128, BATCH * NH), 256, 0, stream>>>(qq, kk, vv, attn);

  gemm_kernel<<<gg, 256, 0, stream>>>(attn, wbf + 3 * C * C, b3, nullptr, x, out, 1);
}
